// Qwen3NextSparseMoeBlock_86449101734227
// MI455X (gfx1250) — matrix operand order repaired and checked by reading, not compiled, not run on hardware
//
#include <hip/hip_runtime.h>
#include <hip/hip_bf16.h>
#include <stdint.h>

// Problem constants (Qwen3-Next MoE block)
#define TT   4096   // B*S tokens
#define DD   2048   // hidden dim
#define MM   512    // expert mlp dim
#define EE   8      // experts
#define AUXW 0.01f
#define AST  40     // padded LDS stride (shorts): 80B -> bank stride 20 dwords

typedef __attribute__((ext_vector_type(16))) __bf16 v16bf;
typedef __attribute__((ext_vector_type(8)))  float  v8f;

// ---------------- helpers ----------------

__device__ __forceinline__ float fast_rcp(float x) {
  return __builtin_amdgcn_rcpf(x);    // v_rcp_f32, 1 ulp
}

__device__ __forceinline__ unsigned short f2bf(float f) {
  union { float f; unsigned u; } c; c.f = f;
  unsigned u = c.u;
  unsigned r = u + 0x7FFFu + ((u >> 16) & 1u);   // round-to-nearest-even
  return (unsigned short)(r >> 16);
}

__device__ __forceinline__ uint4 pack8f(const float* f) {
  uint4 r;
  r.x = (unsigned)f2bf(f[0]) | ((unsigned)f2bf(f[1]) << 16);
  r.y = (unsigned)f2bf(f[2]) | ((unsigned)f2bf(f[3]) << 16);
  r.z = (unsigned)f2bf(f[4]) | ((unsigned)f2bf(f[5]) << 16);
  r.w = (unsigned)f2bf(f[6]) | ((unsigned)f2bf(f[7]) << 16);
  return r;
}

// A-frag 16x32 bf16 from row-major LDS tile (stride AST shorts).
// Lane (m = lane&15, h = lane>>4) owns K = 8h..8h+7 (VGPR 0-3) and
// K = 16+8h..16+8h+7 (VGPR 4-7): two contiguous 16B runs -> 2x ds_load_b128.
__device__ __forceinline__ v16bf load_A(const unsigned short* lds, int rbase) {
  int lane = threadIdx.x & 31;
  const unsigned short* p = lds + (rbase + (lane & 15)) * AST + ((lane >> 4) * 8);
  union { v16bf v; uint4 q[2]; } r;
  r.q[0] = *(const uint4*)p;          // K base .. +7
  r.q[1] = *(const uint4*)(p + 16);   // K base+16 .. +23
  return r.v;
}

// B-frag 32x16 bf16 (KxN) from COLUMN-major LDS tile (B[n*AST + k]).
// Lane (n = lane&15, h = lane>>4) owns column n, K = 8h..8h+7 and
// K = 16+8h..16+8h+7: two contiguous 16B runs -> 2x ds_load_b128.
__device__ __forceinline__ v16bf load_B(const unsigned short* lds, int cbase) {
  int lane = threadIdx.x & 31;
  const unsigned short* p = lds + (cbase + (lane & 15)) * AST + ((lane >> 4) * 8);
  union { v16bf v; uint4 q[2]; } r;
  r.q[0] = *(const uint4*)p;
  r.q[1] = *(const uint4*)(p + 16);
  return r.v;
}

__device__ __forceinline__ v8f wmma_bf16(v16bf a, v16bf b, v8f c) {
  return __builtin_amdgcn_wmma_f32_16x16x32_bf16(false, a, false, b, (short)0, c,
                                                 false, false);
}

#define V8F_ZERO {0.f,0.f,0.f,0.f,0.f,0.f,0.f,0.f}

// ---------------- kernel 0: zero stats ----------------
__global__ void zero_stats_kernel(float* stats) {
  if (threadIdx.x < 16) stats[threadIdx.x] = 0.f;
}

// ---------------- kernel 1: f32 -> bf16 convert of x ----------------
__global__ __launch_bounds__(256) void cvt_x_kernel(const float* __restrict__ x,
                                                    unsigned short* __restrict__ xbf) {
  int idx = blockIdx.x * 256 + threadIdx.x;          // one float4 each
  if (idx >= (TT * DD) / 4) return;
  float4 v = ((const float4*)x)[idx];
  uint2 p;
  p.x = (unsigned)f2bf(v.x) | ((unsigned)f2bf(v.y) << 16);
  p.y = (unsigned)f2bf(v.z) | ((unsigned)f2bf(v.w) << 16);
  ((uint2*)xbf)[idx] = p;
}

// ---------------- kernel 2: router (one wave per token) ----------------
__global__ __launch_bounds__(256) void router_kernel(
    const float* __restrict__ x, const float* __restrict__ wr,
    const float* __restrict__ wg, float* __restrict__ combine,
    float* __restrict__ gate, float* __restrict__ stats) {
  int wave = threadIdx.x >> 5;
  int lane = threadIdx.x & 31;
  int t = blockIdx.x * 8 + wave;

  float acc[8] = {0.f, 0.f, 0.f, 0.f, 0.f, 0.f, 0.f, 0.f};
  float ag = 0.f;
  const float* xr = x + (size_t)t * DD;
  for (int d = lane; d < DD; d += 32) {
    float xv = xr[d];
    const float4* w4 = (const float4*)(wr + (size_t)d * EE);
    float4 w0 = w4[0], w1 = w4[1];
    acc[0] += xv * w0.x; acc[1] += xv * w0.y; acc[2] += xv * w0.z; acc[3] += xv * w0.w;
    acc[4] += xv * w1.x; acc[5] += xv * w1.y; acc[6] += xv * w1.z; acc[7] += xv * w1.w;
    ag += xv * wg[d];
  }
#pragma unroll
  for (int off = 16; off > 0; off >>= 1) {
#pragma unroll
    for (int e = 0; e < 8; ++e) acc[e] += __shfl_down(acc[e], off, 32);
    ag += __shfl_down(ag, off, 32);
  }
  if (lane == 0) {
    float mx = acc[0];
#pragma unroll
    for (int e = 1; e < 8; ++e) mx = fmaxf(mx, acc[e]);
    float p[8], s = 0.f;
#pragma unroll
    for (int e = 0; e < 8; ++e) { p[e] = __expf(acc[e] - mx); s += p[e]; }
    float inv = fast_rcp(s);
#pragma unroll
    for (int e = 0; e < 8; ++e) p[e] *= inv;
    // top-2 (ties -> lowest index, matching lax.top_k)
    int i1 = 0;
#pragma unroll
    for (int e = 1; e < 8; ++e) if (p[e] > p[i1]) i1 = e;
    int i2 = (i1 == 0) ? 1 : 0;
#pragma unroll
    for (int e = 0; e < 8; ++e) if (e != i1 && p[e] > p[i2]) i2 = e;
    float winv = fast_rcp(p[i1] + p[i2]);
#pragma unroll
    for (int e = 0; e < 8; ++e)
      combine[(size_t)t * EE + e] =
          (e == i1) ? p[i1] * winv : ((e == i2) ? p[i2] * winv : 0.f);
    gate[t] = fast_rcp(1.f + __expf(-ag));
#pragma unroll
    for (int e = 0; e < 8; ++e) atomicAdd(&stats[e], p[e]);   // density proxy sum
    atomicAdd(&stats[8 + i1], 1.f);                           // density count
    atomicAdd(&stats[8 + i2], 1.f);
  }
}

// ---------------- kernel 3: H_z = scale * silu(X@W0) * (X@W1), bf16 out -----
// grid: (MM/64, TT/128, 9). z<8: routed expert (scale=combine), z==8: shared
// expert (scale=sigmoid gate). 128x64 tile, 8 waves of 32x32 -> 8 wmma/wave/step.
__global__ __launch_bounds__(256) void h_kernel(
    const unsigned short* __restrict__ xbf, const float* __restrict__ wi0,
    const float* __restrict__ wi1, const float* __restrict__ swi0,
    const float* __restrict__ swi1, const float* __restrict__ combine,
    const float* __restrict__ gate, unsigned short* __restrict__ hb) {
  __shared__ unsigned short Al[128 * AST];     // 128 tokens x 32 k, row-major
  __shared__ unsigned short B0[64 * AST];      // 64 n x 32 k, COLUMN-major
  __shared__ unsigned short B1[64 * AST];

  int z = blockIdx.z;
  int m0 = blockIdx.x * 64;
  int t0 = blockIdx.y * 128;
  const float* w0 = (z < 8) ? wi0 + (size_t)z * DD * MM : swi0;
  const float* w1 = (z < 8) ? wi1 + (size_t)z * DD * MM : swi1;

  int tid = threadIdx.x;
  int wv = tid >> 5, lane = tid & 31;
  int wrow = (wv & 3) * 32, wcol = (wv >> 2) * 32;  // wave -> 32 rows x 32 cols
  int rowX = tid >> 1, cgX = (tid & 1) * 16;        // A staging: 128 x 32
  int nB = tid & 63, kgB = (tid >> 6) * 8;          // B staging: column of 8 k

  v8f a0[2][2] = {{V8F_ZERO, V8F_ZERO}, {V8F_ZERO, V8F_ZERO}};
  v8f a1[2][2] = {{V8F_ZERO, V8F_ZERO}, {V8F_ZERO, V8F_ZERO}};

  for (int k0 = 0; k0 < DD; k0 += 32) {
    // global fetch (before barrier -> overlaps prior wmma)
    const unsigned short* xs = xbf + (size_t)(t0 + rowX) * DD + k0 + cgX;
    uint4 xa = *(const uint4*)xs;
    uint4 xb = *(const uint4*)(xs + 8);
    float f0[8], f1[8];
#pragma unroll
    for (int i = 0; i < 8; ++i) {   // coalesced across nB
      f0[i] = w0[(size_t)(k0 + kgB + i) * MM + m0 + nB];
      f1[i] = w1[(size_t)(k0 + kgB + i) * MM + m0 + nB];
    }
    __syncthreads();
    *(uint4*)&Al[rowX * AST + cgX]     = xa;
    *(uint4*)&Al[rowX * AST + cgX + 8] = xb;
    *(uint4*)&B0[nB * AST + kgB] = pack8f(f0);   // transposed store
    *(uint4*)&B1[nB * AST + kgB] = pack8f(f1);
    __syncthreads();

    v16bf af[2]  = {load_A(Al, wrow), load_A(Al, wrow + 16)};
    v16bf b0f[2] = {load_B(B0, wcol), load_B(B0, wcol + 16)};
    v16bf b1f[2] = {load_B(B1, wcol), load_B(B1, wcol + 16)};
#pragma unroll
    for (int ia = 0; ia < 2; ++ia)
#pragma unroll
      for (int jb = 0; jb < 2; ++jb) {
        a0[ia][jb] = wmma_bf16(af[ia], b0f[jb], a0[ia][jb]);
        a1[ia][jb] = wmma_bf16(af[ia], b1f[jb], a1[ia][jb]);
      }
  }

  // epilogue: scale * silu(a0) * a1, store bf16 to H[z][t][m]
  int nl = lane & 15, hh = (lane >> 4) * 8;
#pragma unroll
  for (int ia = 0; ia < 2; ++ia)
#pragma unroll
    for (int jb = 0; jb < 2; ++jb)
#pragma unroll
      for (int r = 0; r < 8; ++r) {
        int t = t0 + wrow + ia * 16 + r + hh;
        float sc = (z < 8) ? combine[(size_t)t * EE + z] : gate[t];
        float v0 = a0[ia][jb][r], v1 = a1[ia][jb][r];
        float hv = v0 * fast_rcp(1.f + __expf(-v0)) * v1 * sc;   // silu, no div
        hb[((size_t)z * TT + t) * MM + m0 + wcol + jb * 16 + nl] = f2bf(hv);
      }
}

// ---------------- kernel 4: out = H_routed @ wo_flat + H_shared @ sw_wo -----
// Single GEMM, K = E*MM + MM = 4608 (wo is contiguous [E*MM, DD]).
// grid: (DD/64, TT/128). 128x64 tile, 8 waves of 32x32.
__global__ __launch_bounds__(256) void out_kernel(
    const unsigned short* __restrict__ hb, const float* __restrict__ wo,
    const float* __restrict__ swo, float* __restrict__ out) {
  __shared__ unsigned short Al[128 * AST];
  __shared__ unsigned short Bl[64 * AST];      // column-major

  int d0 = blockIdx.x * 64;
  int t0 = blockIdx.y * 128;
  int tid = threadIdx.x;
  int wv = tid >> 5, lane = tid & 31;
  int wrow = (wv & 3) * 32, wcol = (wv >> 2) * 32;
  int rowA = tid >> 1, cgA = (tid & 1) * 16;
  int nB = tid & 63, kgB = (tid >> 6) * 8;

  v8f acc[2][2] = {{V8F_ZERO, V8F_ZERO}, {V8F_ZERO, V8F_ZERO}};

  const int NSTEP = (EE * MM + MM) / 32;  // 144, never crosses an expert slab
  for (int s = 0; s < NSTEP; ++s) {
    int k = s * 32;
    int slab, mo;
    const float* bsrc;
    if (k < EE * MM) { slab = k >> 9; mo = k & 511; bsrc = wo + (size_t)k * DD; }
    else            { slab = 8; mo = k - EE * MM; bsrc = swo + (size_t)(k - EE * MM) * DD; }
    const unsigned short* as =
        hb + ((size_t)slab * TT + t0 + rowA) * MM + mo + cgA;
    uint4 av = *(const uint4*)as;
    uint4 aw = *(const uint4*)(as + 8);
    float fb[8];
#pragma unroll
    for (int i = 0; i < 8; ++i) fb[i] = bsrc[(size_t)(kgB + i) * DD + d0 + nB];
    __syncthreads();
    *(uint4*)&Al[rowA * AST + cgA]     = av;
    *(uint4*)&Al[rowA * AST + cgA + 8] = aw;
    *(uint4*)&Bl[nB * AST + kgB] = pack8f(fb);
    __syncthreads();

    v16bf af[2] = {load_A(Al, wrow), load_A(Al, wrow + 16)};
    v16bf bf[2] = {load_B(Bl, wcol), load_B(Bl, wcol + 16)};
#pragma unroll
    for (int ia = 0; ia < 2; ++ia)
#pragma unroll
      for (int jb = 0; jb < 2; ++jb)
        acc[ia][jb] = wmma_bf16(af[ia], bf[jb], acc[ia][jb]);
  }

  int nl = lane & 15, hh = (lane >> 4) * 8;
#pragma unroll
  for (int ia = 0; ia < 2; ++ia)
#pragma unroll
    for (int jb = 0; jb < 2; ++jb)
#pragma unroll
      for (int r = 0; r < 8; ++r)
        out[(size_t)(t0 + wrow + ia * 16 + r + hh) * DD + d0 + wcol + jb * 16 + nl] =
            acc[ia][jb][r];
}

// ---------------- kernel 5: aux loss ----------------
__global__ void loss_kernel(const float* __restrict__ stats, float* __restrict__ out) {
  // loss = mean_e(density * proxy) * E^2 * AUXW = AUXW*E*sum(cnt*psum)/T^2
  float s = 0.f;
#pragma unroll
  for (int e = 0; e < 8; ++e) s += stats[8 + e] * stats[e];
  out[(size_t)TT * DD] = AUXW * (float)EE * s / ((float)TT * (float)TT);
}

// ---------------- launcher ----------------
extern "C" void kernel_launch(void* const* d_in, const int* in_sizes, int n_in,
                              void* d_out, int out_size, void* d_ws, size_t ws_size,
                              hipStream_t stream) {
  (void)in_sizes; (void)n_in; (void)out_size;
  const float* x    = (const float*)d_in[0];
  const float* wr   = (const float*)d_in[1];
  const float* wi0  = (const float*)d_in[2];
  const float* wi1  = (const float*)d_in[3];
  const float* wo   = (const float*)d_in[4];   // [E*M, D] contiguous
  const float* swi0 = (const float*)d_in[5];
  const float* swi1 = (const float*)d_in[6];
  const float* swo  = (const float*)d_in[7];
  const float* wg   = (const float*)d_in[8];
  float* out = (float*)d_out;

  // workspace layout (bytes)
  char* ws = (char*)d_ws;
  unsigned short* xbf     = (unsigned short*)(ws);                       // 16.8 MB
  unsigned short* hb      = (unsigned short*)(ws + 16777216);            // 37.7 MB
  float*          combine = (float*)(ws + 54525952);                     // 128 KB
  float*          gate    = (float*)(ws + 54657024);                     // 16 KB
  float*          stats   = (float*)(ws + 54673408);                     // 64 B
  if (ws_size < 54673472) return;

  zero_stats_kernel<<<1, 32, 0, stream>>>(stats);
  cvt_x_kernel<<<(TT * DD / 4 + 255) / 256, 256, 0, stream>>>(x, xbf);
  router_kernel<<<TT / 8, 256, 0, stream>>>(x, wr, wg, combine, gate, stats);
  h_kernel<<<dim3(MM / 64, TT / 128, 9), 256, 0, stream>>>(xbf, wi0, wi1, swi0, swi1,
                                                           combine, gate, hb);
  out_kernel<<<dim3(DD / 64, TT / 128), 256, 0, stream>>>(hb, wo, swo, out);
  loss_kernel<<<1, 1, 0, stream>>>(stats, out);
}
